// EncoderLayer_28690381537930
// MI455X (gfx1250) — matrix operand order repaired and checked by reading, not compiled, not run on hardware
//
#include <hip/hip_runtime.h>
#include <math.h>
#include <cstdint>

// ---------------- CDNA5 (gfx1250) types ----------------
typedef __attribute__((ext_vector_type(16))) __bf16 v16bf;
typedef __attribute__((ext_vector_type(8)))  float  v8f;

#define LN_EPS 1e-5f
constexpr int Bn  = 8;
constexpr int Sn  = 1024;
constexpr int Dn  = 1024;
constexpr int Hn  = 16;
constexpr int DKn = 64;
constexpr int DFn = 4096;
constexpr int Mrows = Bn * Sn;   // 8192

__device__ __forceinline__ unsigned short f2bf(float f) {
    unsigned int u = __float_as_uint(f);
    unsigned int r = u + 0x7FFFu + ((u >> 16) & 1u);   // round-to-nearest-even
    return (unsigned short)(r >> 16);
}

union ABfrag { v16bf v; uint4 q[2]; };

// ---------------- generic bf16 WMMA GEMM ----------------
// C[M,N] = A[M,K] @ B[N,K]^T   (+ epilogue)
// A: bf16 (AFMT=0, staged via async-to-LDS DMA) or fp32 converted on the fly (AFMT=1).
// Batched via blockIdx.z: batch = b*Hinner + h; per-operand (outer,inner) strides.
struct GemmP {
    const void*           A;
    const unsigned short* Bm;
    const float*          bias;      // indexed by global n (nullable)
    float*                Cf;        // fp32 out (EPI 2,3)
    unsigned short*       Cb;        // bf16 out (EPI 0,1,4)
    const int*            mask;      // EPI 2
    const float*          residual;  // EPI 3 (fp32, same layout as C)
    const float*          bng; const float* bnb; const float* bnm; const float* bnv;
    long aOuter, aInner, bOuter, bInner, cOuter, cInner;
    int  lda, ldb, ldc, M, N, K, Hinner;
    float alpha;
};

// EPI: 0 = bias -> bf16 store
//      1 = bias -> bf16 store TRANSPOSED as vT[B,H,DK,S]
//      2 = alpha-scale + mask(-1e9) -> fp32 store (scores)
//      3 = bias + residual + batchnorm(n) -> fp32 store
//      4 = bias + exact GELU + batchnorm(n) -> bf16 store
constexpr int BUFE = 128 * 40;   // ushorts per LDS tile buffer (row stride 40, 16B aligned)

template<int AFMT, int EPI, bool GUARD>
__global__ __launch_bounds__(256) void gemm_wmma_kernel(GemmP p) {
    __shared__ unsigned short As[2 * BUFE];
    __shared__ unsigned short Bs[2 * BUFE];

    const int tid  = threadIdx.x;
    const int lane = tid & 31;
    const int wid  = tid >> 5;
    const int wm   = wid >> 2;      // 0..1  (64-row wave tile)
    const int wn   = wid & 3;       // 0..3  (32-col wave tile)

    const int z  = blockIdx.z;
    const int bb = z / p.Hinner;
    const int hh = z % p.Hinner;
    const long baseA = (long)bb * p.aOuter + (long)hh * p.aInner;
    const long baseB = (long)bb * p.bOuter + (long)hh * p.bInner;
    const long baseC = (long)bb * p.cOuter + (long)hh * p.cInner;

    const int m0 = blockIdx.y * 128;
    const int n0 = blockIdx.x * 128;

    v8f acc[4][2];
    #pragma unroll
    for (int i = 0; i < 4; ++i)
        #pragma unroll
        for (int j = 0; j < 2; ++j)
            #pragma unroll
            for (int e = 0; e < 8; ++e) acc[i][j][e] = 0.0f;

    const int kTiles = p.K >> 5;

    // ---- tile staging: 256 thr x 2 chunks of 8 bf16 each for A and for B ----
    auto stageTile = [&](int kt, int buf) {
        const int k0 = kt << 5;
        #pragma unroll
        for (int it = 0; it < 2; ++it) {
            const int c   = tid + it * 256;
            const int row = c >> 2;
            const int col = (c & 3) * 8;
            const int mg  = m0 + row;
            const int ng  = n0 + row;
            if (AFMT == 0) {
                // CDNA5 async DMA: global -> LDS, tracked by ASYNCcnt
                const unsigned short* ap = (const unsigned short*)p.A;
                const unsigned long long ga =
                    (unsigned long long)(uintptr_t)(ap + baseA + (size_t)mg * p.lda + k0 + col);
                const unsigned lofA = (unsigned)(uintptr_t)&As[buf * BUFE + row * 40 + col];
                asm volatile("global_load_async_to_lds_b128 %0, %1, off"
                             :: "v"(lofA), "v"(ga) : "memory");
                const unsigned long long gb =
                    (unsigned long long)(uintptr_t)(p.Bm + baseB + (size_t)ng * p.ldb + k0 + col);
                const unsigned lofB = (unsigned)(uintptr_t)&Bs[buf * BUFE + row * 40 + col];
                asm volatile("global_load_async_to_lds_b128 %0, %1, off"
                             :: "v"(lofB), "v"(gb) : "memory");
            } else {
                // fp32 A -> convert to bf16 in LDS
                const float* ap  = (const float*)p.A;
                const float* src = ap + baseA + (size_t)mg * p.lda + k0 + col;
                const float4 f0 = *(const float4*)(src);
                const float4 f1 = *(const float4*)(src + 4);
                union { uint4 q; unsigned short h[8]; } pk;
                pk.h[0] = f2bf(f0.x); pk.h[1] = f2bf(f0.y);
                pk.h[2] = f2bf(f0.z); pk.h[3] = f2bf(f0.w);
                pk.h[4] = f2bf(f1.x); pk.h[5] = f2bf(f1.y);
                pk.h[6] = f2bf(f1.z); pk.h[7] = f2bf(f1.w);
                *(uint4*)(&As[buf * BUFE + row * 40 + col]) = pk.q;
                if (!GUARD || ng < p.N) {
                    const uint4 vb = *(const uint4*)(p.Bm + baseB + (size_t)ng * p.ldb + k0 + col);
                    *(uint4*)(&Bs[buf * BUFE + row * 40 + col]) = vb;
                }
            }
        }
    };

    stageTile(0, 0);
    for (int kt = 0; kt < kTiles; ++kt) {
        if (kt + 1 < kTiles) {
            stageTile(kt + 1, (kt + 1) & 1);
            if (AFMT == 0)
                asm volatile("s_wait_asynccnt 0x4" ::: "memory");  // tile kt landed (in-order)
        } else {
            if (AFMT == 0)
                asm volatile("s_wait_asynccnt 0x0" ::: "memory");
        }
        __syncthreads();

        // ---- load fragments per CDNA5 16-bit WMMA layouts ----
        const int buf = kt & 1;
        ABfrag afr[4], bfr[2];
        const int ml = lane & 15;
        const int kh = lane >> 4;
        #pragma unroll
        for (int i = 0; i < 4; ++i) {
            const unsigned short* pA = &As[buf * BUFE + (wm * 64 + i * 16 + ml) * 40];
            afr[i].q[0] = *(const uint4*)(pA + kh * 8);        // K = kh*8 .. +7
            afr[i].q[1] = *(const uint4*)(pA + 16 + kh * 8);   // K = 16+kh*8 .. +7
        }
        #pragma unroll
        for (int j = 0; j < 2; ++j) {
            const unsigned short* pB = &Bs[buf * BUFE + (wn * 32 + j * 16 + ml) * 40];
            bfr[j].q[0] = *(const uint4*)(pB + kh * 8);        // K = kh*8 .. +7
            bfr[j].q[1] = *(const uint4*)(pB + 16 + kh * 8);   // K = 16+kh*8 .. +7
        }
        #pragma unroll
        for (int i = 0; i < 4; ++i)
            #pragma unroll
            for (int j = 0; j < 2; ++j)
                acc[i][j] = __builtin_amdgcn_wmma_f32_16x16x32_bf16(
                    false, afr[i].v, false, bfr[j].v, (short)0, acc[i][j], false, false);
        __syncthreads();
    }

    // ---- epilogue: C layout lane n = lane%16, row = r + 8*(lane/16) ----
    const int mlane = lane & 15;
    const int hl    = lane >> 4;
    #pragma unroll
    for (int i = 0; i < 4; ++i) {
        #pragma unroll
        for (int j = 0; j < 2; ++j) {
            #pragma unroll
            for (int r = 0; r < 8; ++r) {
                const int mg = m0 + wm * 64 + i * 16 + hl * 8 + r;
                const int ng = n0 + wn * 32 + j * 16 + mlane;
                if (GUARD) { if (mg >= p.M || ng >= p.N) continue; }
                float val = acc[i][j][r];
                if (EPI == 0 || EPI == 1 || EPI == 3 || EPI == 4) {
                    if (p.bias) val += p.bias[ng];
                }
                if (EPI == 0) {
                    p.Cb[baseC + (size_t)mg * p.ldc + ng] = f2bf(val);
                } else if (EPI == 1) {   // vT[b][h][dk][s]
                    const int h2 = ng >> 6, nl = ng & 63;
                    const int b2 = mg >> 10, ms = mg & (Sn - 1);
                    const size_t idx = (((size_t)b2 * Hn + h2) * DKn + nl) * Sn + ms;
                    p.Cb[idx] = f2bf(val);
                } else if (EPI == 2) {
                    val *= p.alpha;
                    if (p.mask[(size_t)bb * Sn + ng] == 0) val = -1e9f;
                    p.Cf[baseC + (size_t)mg * p.ldc + ng] = val;
                } else if (EPI == 3) {
                    val += p.residual[(size_t)mg * p.ldc + ng];
                    const float sc = p.bng[ng] * rsqrtf(p.bnv[ng] + LN_EPS);
                    val = (val - p.bnm[ng]) * sc + p.bnb[ng];
                    p.Cf[baseC + (size_t)mg * p.ldc + ng] = val;
                } else if (EPI == 4) {
                    val = 0.5f * val * (1.0f + erff(val * 0.70710678118654752f));
                    const float sc = p.bng[ng] * rsqrtf(p.bnv[ng] + LN_EPS);
                    val = (val - p.bnm[ng]) * sc + p.bnb[ng];
                    p.Cb[baseC + (size_t)mg * p.ldc + ng] = f2bf(val);
                }
            }
        }
    }
}

// ---------------- LayerNorm (row of 1024) -> bf16 (+ optional raw bf16 copy) ----------------
__global__ __launch_bounds__(256) void ln_kernel(const float* __restrict__ x,
                                                 const float* __restrict__ g,
                                                 const float* __restrict__ b,
                                                 unsigned short* __restrict__ outb,
                                                 unsigned short* __restrict__ rawb) {
    const size_t row = blockIdx.x;
    const float* xr = x + row * Dn;
    __shared__ float red[256];
    const int tid = threadIdx.x;
    float v[4]; float s = 0.0f;
    #pragma unroll
    for (int i = 0; i < 4; ++i) { v[i] = xr[tid + i * 256]; s += v[i]; }
    red[tid] = s; __syncthreads();
    for (int o = 128; o > 0; o >>= 1) { if (tid < o) red[tid] += red[tid + o]; __syncthreads(); }
    const float mu = red[0] * (1.0f / Dn);
    __syncthreads();
    float s2 = 0.0f;
    #pragma unroll
    for (int i = 0; i < 4; ++i) { const float d = v[i] - mu; s2 += d * d; }
    red[tid] = s2; __syncthreads();
    for (int o = 128; o > 0; o >>= 1) { if (tid < o) red[tid] += red[tid + o]; __syncthreads(); }
    const float rstd = rsqrtf(red[0] * (1.0f / Dn) + LN_EPS);
    #pragma unroll
    for (int i = 0; i < 4; ++i) {
        const int c = tid + i * 256;
        const float y = (v[i] - mu) * rstd * g[c] + b[c];
        outb[row * Dn + c] = f2bf(y);
        if (rawb) rawb[row * Dn + c] = f2bf(v[i]);
    }
}

// ---------------- row softmax, in place, rows of 1024 fp32 ----------------
__global__ __launch_bounds__(256) void softmax_kernel(float* __restrict__ a) {
    const size_t row = blockIdx.x;
    float* r = a + row * 1024;
    __shared__ float red[256];
    const int tid = threadIdx.x;
    float v[4]; float mx = -3.4e38f;
    #pragma unroll
    for (int i = 0; i < 4; ++i) { v[i] = r[tid + i * 256]; mx = fmaxf(mx, v[i]); }
    red[tid] = mx; __syncthreads();
    for (int o = 128; o > 0; o >>= 1) { if (tid < o) red[tid] = fmaxf(red[tid], red[tid + o]); __syncthreads(); }
    mx = red[0]; __syncthreads();
    float s = 0.0f;
    #pragma unroll
    for (int i = 0; i < 4; ++i) { v[i] = __expf(v[i] - mx); s += v[i]; }
    red[tid] = s; __syncthreads();
    for (int o = 128; o > 0; o >>= 1) { if (tid < o) red[tid] += red[tid + o]; __syncthreads(); }
    const float inv = 1.0f / red[0];
    #pragma unroll
    for (int i = 0; i < 4; ++i) r[tid + i * 256] = v[i] * inv;
}

// ---------------- fp32 -> bf16 convert ----------------
__global__ __launch_bounds__(256) void convert_bf16(const float* __restrict__ in,
                                                    unsigned short* __restrict__ out, int n) {
    const int base = blockIdx.x * 1024 + threadIdx.x;
    #pragma unroll
    for (int it = 0; it < 4; ++it) {
        const int idx = base + it * 256;
        if (idx < n) out[idx] = f2bf(in[idx]);
    }
}

// ---------------- host orchestration ----------------
extern "C" void kernel_launch(void* const* d_in, const int* in_sizes, int n_in,
                              void* d_out, int out_size, void* d_ws, size_t ws_size,
                              hipStream_t stream) {
    (void)in_sizes; (void)n_in; (void)out_size; (void)ws_size;
    const float* x        = (const float*)d_in[0];
    const int*   mask     = (const int*)  d_in[1];
    const float* Wq       = (const float*)d_in[2];  const float* bq = (const float*)d_in[3];
    const float* Wk       = (const float*)d_in[4];  const float* bk = (const float*)d_in[5];
    const float* Wv       = (const float*)d_in[6];  const float* bv = (const float*)d_in[7];
    const float* Wo       = (const float*)d_in[8];  const float* bo = (const float*)d_in[9];
    const float* ln_a_g   = (const float*)d_in[10]; const float* ln_a_b = (const float*)d_in[11];
    const float* W1       = (const float*)d_in[12]; const float* b1 = (const float*)d_in[13];
    const float* W2       = (const float*)d_in[14]; const float* b2 = (const float*)d_in[15];
    const float* ln_f_g   = (const float*)d_in[16]; const float* ln_f_b = (const float*)d_in[17];
    const float* bnff_g   = (const float*)d_in[18]; const float* bnff_b = (const float*)d_in[19];
    const float* bnff_m   = (const float*)d_in[20]; const float* bnff_v = (const float*)d_in[21];
    const float* bn1_g    = (const float*)d_in[22]; const float* bn1_b  = (const float*)d_in[23];
    const float* bn1_m    = (const float*)d_in[24]; const float* bn1_v  = (const float*)d_in[25];
    const float* bn2_g    = (const float*)d_in[26]; const float* bn2_b  = (const float*)d_in[27];
    const float* bn2_m    = (const float*)d_in[28]; const float* bn2_v  = (const float*)d_in[29];

    char* ws = (char*)d_ws;
    const size_t MB_ = 1024ull * 1024ull;
    unsigned short* x_bf  = (unsigned short*)(ws +   0 * MB_);
    unsigned short* xq_bf = (unsigned short*)(ws +  16 * MB_);
    unsigned short* Wq_bf = (unsigned short*)(ws +  32 * MB_);
    unsigned short* Wk_bf = (unsigned short*)(ws +  34 * MB_);
    unsigned short* Wv_bf = (unsigned short*)(ws +  36 * MB_);
    unsigned short* Wo_bf = (unsigned short*)(ws +  38 * MB_);
    unsigned short* W1_bf = (unsigned short*)(ws +  40 * MB_);
    unsigned short* W2_bf = (unsigned short*)(ws +  48 * MB_);
    unsigned short* q_bf  = (unsigned short*)(ws +  56 * MB_);
    unsigned short* k_bf  = (unsigned short*)(ws +  72 * MB_);
    unsigned short* vT_bf = (unsigned short*)(ws +  88 * MB_);
    float*          x1    = (float*)         (ws + 104 * MB_);
    unsigned short* h_bf  = (unsigned short*)(ws + 136 * MB_);
    unsigned short* ctx_bf = q_bf;   // q dead after scores
    unsigned short* xn_bf  = k_bf;   // k dead after scores

    float* out_x = (float*)d_out;
    float* attn  = out_x + (size_t)Bn * Sn * Dn;   // [B,H,S,S] region of d_out

    const dim3 blk(256);
    // weight conversion
    convert_bf16<<<dim3((Dn*Dn   + 1023) / 1024), blk, 0, stream>>>(Wq, Wq_bf, Dn*Dn);
    convert_bf16<<<dim3((Dn*Dn   + 1023) / 1024), blk, 0, stream>>>(Wk, Wk_bf, Dn*Dn);
    convert_bf16<<<dim3((Dn*Dn   + 1023) / 1024), blk, 0, stream>>>(Wv, Wv_bf, Dn*Dn);
    convert_bf16<<<dim3((Dn*Dn   + 1023) / 1024), blk, 0, stream>>>(Wo, Wo_bf, Dn*Dn);
    convert_bf16<<<dim3((DFn*Dn  + 1023) / 1024), blk, 0, stream>>>(W1, W1_bf, DFn*Dn);
    convert_bf16<<<dim3((Dn*DFn  + 1023) / 1024), blk, 0, stream>>>(W2, W2_bf, Dn*DFn);

    // pre-LN for q, plus raw x in bf16 for k/v
    ln_kernel<<<dim3(Mrows), blk, 0, stream>>>(x, ln_a_g, ln_a_b, xq_bf, x_bf);

    GemmP p{};
    // ---- q = LN(x) @ Wq^T + bq ----
    p = GemmP{}; p.A = xq_bf; p.Bm = Wq_bf; p.bias = bq; p.Cb = q_bf;
    p.lda = Dn; p.ldb = Dn; p.ldc = Dn; p.M = Mrows; p.N = Dn; p.K = Dn; p.Hinner = 1;
    gemm_wmma_kernel<0,0,false><<<dim3(Dn/128, Mrows/128, 1), blk, 0, stream>>>(p);
    // ---- k = x @ Wk^T + bk ----
    p.A = x_bf; p.Bm = Wk_bf; p.bias = bk; p.Cb = k_bf;
    gemm_wmma_kernel<0,0,false><<<dim3(Dn/128, Mrows/128, 1), blk, 0, stream>>>(p);
    // ---- vT = transpose(x @ Wv^T + bv) -> [B,H,DK,S] ----
    p.A = x_bf; p.Bm = Wv_bf; p.bias = bv; p.Cb = vT_bf;
    gemm_wmma_kernel<0,1,false><<<dim3(Dn/128, Mrows/128, 1), blk, 0, stream>>>(p);

    // ---- scores[b,h] = (q_h @ k_h^T)/8, masked -> fp32 into d_out attn region ----
    p = GemmP{}; p.A = q_bf; p.Bm = k_bf; p.Cf = attn; p.mask = mask;
    p.lda = Dn; p.ldb = Dn; p.ldc = Sn; p.M = Sn; p.N = Sn; p.K = DKn; p.Hinner = Hn;
    p.aOuter = (long)Sn*Dn; p.aInner = DKn;
    p.bOuter = (long)Sn*Dn; p.bInner = DKn;
    p.cOuter = (long)Hn*Sn*Sn; p.cInner = (long)Sn*Sn;
    p.alpha = 0.125f;
    gemm_wmma_kernel<0,2,false><<<dim3(Sn/128, Sn/128, Bn*Hn), blk, 0, stream>>>(p);

    // ---- softmax rows of attn ----
    softmax_kernel<<<dim3(Bn*Hn*Sn), blk, 0, stream>>>(attn);

    // ---- context[b,h] = attn_h @ v_h  (A fp32 from d_out, B = vT) -> ctx bf16 [B*S, D] ----
    p = GemmP{}; p.A = attn; p.Bm = vT_bf; p.Cb = ctx_bf;
    p.lda = Sn; p.ldb = Sn; p.ldc = Dn; p.M = Sn; p.N = DKn; p.K = Sn; p.Hinner = Hn;
    p.aOuter = (long)Hn*Sn*Sn; p.aInner = (long)Sn*Sn;
    p.bOuter = (long)Hn*DKn*Sn; p.bInner = (long)DKn*Sn;
    p.cOuter = (long)Sn*Dn; p.cInner = DKn;
    gemm_wmma_kernel<1,0,true><<<dim3(1, Sn/128, Bn*Hn), blk, 0, stream>>>(p);

    // ---- x1 = bn1(x + ctx @ Wo^T + bo) -> fp32 ----
    p = GemmP{}; p.A = ctx_bf; p.Bm = Wo_bf; p.bias = bo; p.Cf = x1; p.residual = x;
    p.bng = bn1_g; p.bnb = bn1_b; p.bnm = bn1_m; p.bnv = bn1_v;
    p.lda = Dn; p.ldb = Dn; p.ldc = Dn; p.M = Mrows; p.N = Dn; p.K = Dn; p.Hinner = 1;
    gemm_wmma_kernel<0,3,false><<<dim3(Dn/128, Mrows/128, 1), blk, 0, stream>>>(p);

    // ---- xn = LN_ff(x1) ----
    ln_kernel<<<dim3(Mrows), blk, 0, stream>>>(x1, ln_f_g, ln_f_b, xn_bf, (unsigned short*)nullptr);

    // ---- h = bnff(gelu(xn @ W1^T + b1)) -> bf16 ----
    p = GemmP{}; p.A = xn_bf; p.Bm = W1_bf; p.bias = b1; p.Cb = h_bf;
    p.bng = bnff_g; p.bnb = bnff_b; p.bnm = bnff_m; p.bnv = bnff_v;
    p.lda = Dn; p.ldb = Dn; p.ldc = DFn; p.M = Mrows; p.N = DFn; p.K = Dn; p.Hinner = 1;
    gemm_wmma_kernel<0,4,false><<<dim3(DFn/128, Mrows/128, 1), blk, 0, stream>>>(p);

    // ---- out = bn2(x1 + h @ W2^T + b2) -> fp32 d_out ----
    p = GemmP{}; p.A = h_bf; p.Bm = W2_bf; p.bias = b2; p.Cf = out_x; p.residual = x1;
    p.bng = bn2_g; p.bnb = bn2_b; p.bnm = bn2_m; p.bnv = bn2_v;
    p.lda = DFn; p.ldb = DFn; p.ldc = Dn; p.M = Mrows; p.N = Dn; p.K = DFn; p.Hinner = 1;
    gemm_wmma_kernel<0,3,false><<<dim3(Dn/128, Mrows/128, 1), blk, 0, stream>>>(p);
}
